// SAGENet_46488726012188
// MI455X (gfx1250) — matrix operand order repaired and checked by reading, not compiled, not run on hardware
//
#include <hip/hip_runtime.h>
#include <math.h>

// Problem constants (from reference)
#define N0_   260000
#define N1_   10240
#define B_    1024
#define E0_   256000
#define E1_   10240
#define IN_C_ 602
#define HID_  256
#define OUT_C_ 41

// GEMM tiling (bf16 WMMA 16x16x32)
#define K0P     1216   // 2*602=1204 padded to multiple of 32
#define K0STEPS 38
#define K1P     512    // 2*256
#define K1STEPS 16
#define N0TILES 16     // 256/16
#define N1TILES 3      // 41 padded to 48 -> 3 tiles
#define NPAD1   48

typedef __attribute__((ext_vector_type(16))) __bf16       v16bf;
typedef __attribute__((ext_vector_type(8)))  float        v8f;
typedef __attribute__((ext_vector_type(4)))  unsigned int u32x4;

static __device__ __forceinline__ unsigned short f32_to_bf16(float f) {
  union { float f; unsigned u; } v; v.f = f;
  unsigned u = v.u;
  u += 0x7FFFu + ((u >> 16) & 1u);   // round-to-nearest-even
  return (unsigned short)(u >> 16);
}

static __device__ __forceinline__ int lower_bound_i32(const int* a, int n, int val) {
  int lo = 0, hi = n;
  while (lo < hi) { int mid = (lo + hi) >> 1; if (a[mid] < val) lo = mid + 1; else hi = mid; }
  return lo;
}

// ---------------------------------------------------------------------------
// Pack stacked weights [Wl; Wr] (fp32 row-major, each Kl x Ncols) into the
// per-lane bf16 WMMA B-fragment layout: fragment index = n_tile*ksteps + ks,
// lane l holds column n = n0 + l%16, K = ks*32 + (l>>4)*8 + i (i<8), +16+(i-8) (i>=8).
// Rows >= 2*Kl and cols >= Ncols are zero padding.
// ---------------------------------------------------------------------------
__global__ void pack_weights(const float* __restrict__ Wl, const float* __restrict__ Wr,
                             int Kl, int Ncols, int ksteps,
                             unsigned short* __restrict__ Bpack) {
  int lane = threadIdx.x & 31;
  int frag = blockIdx.x;                 // n_tile*ksteps + kstep
  int n_tile = frag / ksteps;
  int kstep  = frag - n_tile * ksteps;
  int n  = n_tile * 16 + (lane & 15);
  int kb = kstep * 32 + ((lane >> 4) << 3);
  union { u32x4 q[2]; unsigned short s[16]; } f;
  for (int i = 0; i < 16; ++i) {
    int k = kb + (i < 8 ? i : i + 8);
    float x = 0.f;
    if (n < Ncols) {
      if (k < Kl)           x = Wl[(size_t)k * Ncols + n];
      else if (k < 2 * Kl)  x = Wr[(size_t)(k - Kl) * Ncols + n];
    }
    f.s[i] = f32_to_bf16(x);
  }
  u32x4* dst = (u32x4*)(Bpack + (((size_t)frag * 32 + lane) << 4));
  dst[0] = f.q[0];
  dst[1] = f.q[1];
}

// ---------------------------------------------------------------------------
// Layer-0 segmented mean + bf16 conversion. dst0 is sorted, so target t owns
// the contiguous edge range [lower_bound(t), lower_bound(t+1)). One block of
// 256 threads per target; thread covers cols {c, c+256, c+512} of 602.
// Writes A0 row t = [mean_agg(602) | x_t(602) | 0-pad(12)] as bf16.
// ---------------------------------------------------------------------------
__global__ void agg_layer0(const float* __restrict__ x, const int* __restrict__ src,
                           const int* __restrict__ dst, unsigned short* __restrict__ A0) {
  int t = blockIdx.x;
  int tid = threadIdx.x;
  int lo = lower_bound_i32(dst, E0_, t);
  int hi = lower_bound_i32(dst, E0_, t + 1);
  int c0 = tid, c1 = tid + 256, c2 = tid + 512;
  float a0 = 0.f, a1 = 0.f, a2 = 0.f;
  for (int e = lo; e < hi; ++e) {
    const float* row = x + (size_t)src[e] * IN_C_;
    a0 += row[c0];
    if (c1 < IN_C_) a1 += row[c1];
    if (c2 < IN_C_) a2 += row[c2];
  }
  int cnt = hi - lo;
  float inv = 1.f / (float)(cnt > 0 ? cnt : 1);
  unsigned short* out = A0 + (size_t)t * K0P;
  const float* xt = x + (size_t)t * IN_C_;   // targets are first N1 rows of x
  out[c0] = f32_to_bf16(a0 * inv);
  out[IN_C_ + c0] = f32_to_bf16(xt[c0]);
  if (c1 < IN_C_) { out[c1] = f32_to_bf16(a1 * inv); out[IN_C_ + c1] = f32_to_bf16(xt[c1]); }
  if (c2 < IN_C_) { out[c2] = f32_to_bf16(a2 * inv); out[IN_C_ + c2] = f32_to_bf16(xt[c2]); }
  if (tid < K0P - 2 * IN_C_) out[2 * IN_C_ + tid] = 0;  // K padding
}

// Layer-1 segmented mean over h[N1,256]; writes A1 row t = [mean | h_t] bf16.
__global__ void agg_layer1(const float* __restrict__ h, const int* __restrict__ src,
                           const int* __restrict__ dst, unsigned short* __restrict__ A1) {
  int t = blockIdx.x;
  int c = threadIdx.x;                 // 256 threads = 256 cols
  int lo = lower_bound_i32(dst, E1_, t);
  int hi = lower_bound_i32(dst, E1_, t + 1);
  float a = 0.f;
  for (int e = lo; e < hi; ++e) a += h[(size_t)src[e] * HID_ + c];
  int cnt = hi - lo;
  float inv = 1.f / (float)(cnt > 0 ? cnt : 1);
  unsigned short* out = A1 + (size_t)t * K1P;
  out[c]         = f32_to_bf16(a * inv);
  out[HID_ + c]  = f32_to_bf16(h[(size_t)t * HID_ + c]);
}

// ---------------------------------------------------------------------------
// bf16 WMMA GEMM: C[M x ldc] = A[M x lda(bf16)] * B + bias (+ReLU).
// One wave per 16x16 tile, 8 waves/block, exact tile count (EXEC all-ones).
// A per-lane fragment (ISA 16-bit A 16x32): m = m0 + l%16,
//   elems 0..7 -> K = k0 + (l>>4)*8 + 0..7, elems 8..15 -> +16..23.
// B fragments pre-packed contiguous (32 B/lane) by pack_weights.
// ---------------------------------------------------------------------------
__global__ void gemm_bf16_wmma(const unsigned short* __restrict__ A,
                               const unsigned short* __restrict__ Bpack,
                               const float* __restrict__ bias,
                               float* __restrict__ C,
                               int Ntiles, int ksteps, int lda, int ldc,
                               int Ncols, int relu) {
  int lane = threadIdx.x & 31;
  int wave = threadIdx.x >> 5;
  int tile = blockIdx.x * (blockDim.x >> 5) + wave;
  int m_tile = tile / Ntiles;
  int n_tile = tile - m_tile * Ntiles;
  int m0 = m_tile << 4, n0 = n_tile << 4;

  const unsigned short* pA = A + (size_t)(m0 + (lane & 15)) * lda + ((lane >> 4) << 3);
  const u32x4* pB = (const u32x4*)Bpack + ((size_t)n_tile * ksteps * 32 + lane) * 2;

  union { v8f v; float f[8]; } acc;
  for (int i = 0; i < 8; ++i) acc.f[i] = 0.f;

  for (int ks = 0; ks < ksteps; ++ks) {
    union { v16bf v; u32x4 q[2]; } a, b;
    a.q[0] = *(const u32x4*)(pA);
    a.q[1] = *(const u32x4*)(pA + 16);
    b.q[0] = pB[0];
    b.q[1] = pB[1];
    pA += 32;       // K advances by 32 per step
    pB += 64;       // one fragment = 32 lanes * 32 B = 64 u32x4
    acc.v = __builtin_amdgcn_wmma_f32_16x16x32_bf16(
        /*neg_a=*/false, a.v, /*neg_b=*/false, b.v,
        /*c_mod=*/(short)0, acc.v, /*reuse_a=*/false, /*reuse_b=*/false);
  }

  // D layout: vgpr j -> M = m0 + j + 8*(lane>>4), N = n0 + lane%16
  int n = n0 + (lane & 15);
  float bv = (n < Ncols) ? bias[n] : 0.f;
  int mbase = m0 + ((lane >> 4) << 3);
  for (int j = 0; j < 8; ++j) {
    float v = acc.f[j] + bv;
    if (relu) v = fmaxf(v, 0.f);
    C[(size_t)(mbase + j) * ldc + n] = v;
  }
}

// log-softmax over 41 logits per row (padded rows of 48 in scratch)
__global__ void log_softmax_41(const float* __restrict__ logits, float* __restrict__ out) {
  __shared__ float l[OUT_C_];
  int row = blockIdx.x, tid = threadIdx.x;   // 64 threads
  if (tid < OUT_C_) l[tid] = logits[(size_t)row * NPAD1 + tid];
  __syncthreads();
  if (tid < OUT_C_) {
    float m = -1e30f;
    for (int i = 0; i < OUT_C_; ++i) m = fmaxf(m, l[i]);
    float s = 0.f;
    for (int i = 0; i < OUT_C_; ++i) s += expf(l[i] - m);
    out[(size_t)row * OUT_C_ + tid] = l[tid] - m - logf(s);
  }
}

extern "C" void kernel_launch(void* const* d_in, const int* in_sizes, int n_in,
                              void* d_out, int out_size, void* d_ws, size_t ws_size,
                              hipStream_t stream) {
  (void)in_sizes; (void)n_in; (void)out_size; (void)ws_size;
  const float* x    = (const float*)d_in[0];
  const float* W_l0 = (const float*)d_in[1];
  const float* b_l0 = (const float*)d_in[2];
  const float* W_r0 = (const float*)d_in[3];
  const float* W_l1 = (const float*)d_in[4];
  const float* b_l1 = (const float*)d_in[5];
  const float* W_r1 = (const float*)d_in[6];
  const int* src0 = (const int*)d_in[7];
  const int* dst0 = (const int*)d_in[8];
  const int* src1 = (const int*)d_in[9];
  const int* dst1 = (const int*)d_in[10];
  float* out = (float*)d_out;

  // Workspace carve-out (~37 MB), 256-B aligned slices.
  char* ws = (char*)d_ws;
  size_t off = 0;
  auto carve = [&](size_t bytes) -> char* {
    char* p = ws + off;
    off += (bytes + 255) & ~(size_t)255;
    return p;
  };
  unsigned short* A0 = (unsigned short*)carve((size_t)N1_ * K0P * 2);                 // 24.9 MB
  unsigned short* B0 = (unsigned short*)carve((size_t)N0TILES * K0STEPS * 512 * 2);   // 623 KB
  float*          h  = (float*)carve((size_t)N1_ * HID_ * 4);                         // 10.5 MB
  unsigned short* A1 = (unsigned short*)carve((size_t)B_ * K1P * 2);                  // 1 MB
  unsigned short* B1 = (unsigned short*)carve((size_t)N1TILES * K1STEPS * 512 * 2);   // 48 KB
  float*      logits = (float*)carve((size_t)B_ * NPAD1 * 4);                         // 192 KB

  // Weight packing (tiny, once per launch)
  pack_weights<<<N0TILES * K0STEPS, 32, 0, stream>>>(W_l0, W_r0, IN_C_, HID_,  K0STEPS, B0);
  pack_weights<<<N1TILES * K1STEPS, 32, 0, stream>>>(W_l1, W_r1, HID_,  OUT_C_, K1STEPS, B1);

  // Layer 0: mean-aggregate (bandwidth-bound ~616 MB gather) + fused bf16 cast
  agg_layer0<<<N1_, 256, 0, stream>>>(x, src0, dst0, A0);
  // h = relu([agg|x_tgt] @ [W_l0;W_r0] + b_l0)  -- 640*16 tiles, 8 waves/block
  gemm_bf16_wmma<<<(N1_ / 16) * N0TILES / 8, 256, 0, stream>>>(
      A0, B0, b_l0, h, N0TILES, K0STEPS, K0P, HID_, HID_, 1);

  // Layer 1
  agg_layer1<<<B_, 256, 0, stream>>>(h, src1, dst1, A1);
  gemm_bf16_wmma<<<(B_ / 16) * N1TILES / 8, 256, 0, stream>>>(
      A1, B1, b_l1, logits, N1TILES, K1STEPS, K1P, NPAD1, OUT_C_, 0);

  log_softmax_41<<<B_, 64, 0, stream>>>(logits, out);
}
